// PartDict_77318001263272
// MI455X (gfx1250) — compile-verified, matrix operand order checked by reading, not run on hardware
//
#include <hip/hip_runtime.h>
#include <hip/hip_bf16.h>

typedef __attribute__((ext_vector_type(16))) __bf16 v16bf;
typedef __attribute__((ext_vector_type(8)))  float  v8f;

#define C_DIM   256
#define N_CODES 2048
#define NTILES  128   // 2048/16
#define KBLKS   8     // 256/32
#define HC      28
#define POS     784
#define PTILES  49
#define BATCH   32
#define ROWSTRIDE 2049  // padded dist row stride in dwords (2049%64==1 -> conflict-free columns)

// workspace layout (bytes)
#define WS_BPACK 0
#define WS_E2    (2u*1024u*1024u)
#define WS_BOWP  (WS_E2 + N_CODES*4u)

// K-index inside a 16x32 bf16 WMMA fragment for element j of lane (khalf = lane/16)
__device__ __forceinline__ int klocal(int j, int khalf) {
  return (j < 8) ? (j + 8 * khalf) : (j + 8 + 8 * khalf);
}

// ---------------- kernel 1a: e2[n] = sum_k emb[n][k]^2 ----------------
__global__ __launch_bounds__(256)
void k_e2(const float* __restrict__ emb, float* __restrict__ e2) {
  int n = blockIdx.x * 256 + threadIdx.x;
  if (n < N_CODES) {
    const float* row = emb + (size_t)n * C_DIM;
    float s = 0.f;
    for (int k = 0; k < C_DIM; ++k) { float v = row[k]; s += v * v; }
    e2[n] = s;
  }
}

// ------- kernel 1b: pack embedding into bf16 hi/lo WMMA B fragments -------
// layout: [ntile(128)][kb(8)][part(2)][lane(32)][16 bf16]  (1024 B per fragment)
__global__ __launch_bounds__(256)
void k_pack(const float* __restrict__ emb, unsigned char* __restrict__ bpack) {
  int nt = blockIdx.x;           // 0..127
  int w  = threadIdx.x >> 5;     // kb = 0..7
  int l  = threadIdx.x & 31;
  int n  = nt * 16 + (l & 15);
  int kh = l >> 4;
  const float* row = emb + (size_t)n * C_DIM + w * 32;
  v16bf hi, lo;
#pragma unroll
  for (int j = 0; j < 16; ++j) {
    float x = row[klocal(j, kh)];
    __bf16 h = (__bf16)x;
    hi[j] = h;
    lo[j] = (__bf16)(x - (float)h);
  }
  size_t base = (size_t)(nt * KBLKS + w) * 2048 + (size_t)l * 32;
  *(v16bf*)(bpack + base)        = hi;
  *(v16bf*)(bpack + base + 1024) = lo;
}

// ---------------- kernel 2: fused GEMM + softmax + bow partials ----------------
__global__ __launch_bounds__(256)
void k_main(const float* __restrict__ feat, const float* __restrict__ e2g,
            const unsigned char* __restrict__ bpack, const float* __restrict__ mdm,
            float* __restrict__ codes, float* __restrict__ bowp)
{
  extern __shared__ __align__(32) unsigned char smem[];
  __bf16* s_a   = (__bf16*)smem;                         // 16 KB: A fragments [kb][part][lane][16]
  float*  s_f2p = (float*)(smem + 16384);                // 16x16 partial |f|^2
  float*  s_f2  = (float*)(smem + 16384 + 1024);         // 16
  float*  s_inv = (float*)(smem + 16384 + 1024 + 64);    // 16
  float*  s_d   = (float*)(smem + 16384 + 1024 + 128);   // 16 x 2049 dist/exp tile

  const int pt = blockIdx.x, b = blockIdx.y;
  const int t = threadIdx.x, w = t >> 5, l = t & 31;
  const int m = l & 15, kh = l >> 4;

  // ---- Phase A: stage A fragments (bf16 hi/lo) + |f|^2 partials ----
  {
    const int p = pt * 16 + m;
    const int y = p / HC, x = p % HC;
    const float* fb = feat + (((size_t)b * C_DIM) * 30 + (y + 1)) * 30 + (x + 1);
    v16bf hi, lo; float sq = 0.f;
#pragma unroll
    for (int j = 0; j < 16; ++j) {
      int k = w * 32 + klocal(j, kh);
      float v = fb[(size_t)k * 900];
      sq += v * v;
      __bf16 h = (__bf16)v;
      hi[j] = h;
      lo[j] = (__bf16)(v - (float)h);
    }
    *(v16bf*)(s_a + ((w * 2 + 0) * 32 + l) * 16) = hi;
    *(v16bf*)(s_a + ((w * 2 + 1) * 32 + l) * 16) = lo;
    s_f2p[m * 16 + w * 2 + kh] = sq;
  }
  __syncthreads();
  if (t < 16) {
    float s = 0.f;
#pragma unroll
    for (int i = 0; i < 16; ++i) s += s_f2p[t * 16 + i];
    s_f2[t] = s;
  }
  __syncthreads();

  // ---- Phase B: WMMA sweep over all 128 N-tiles (16 per wave) ----
  v16bf a[KBLKS][2];
#pragma unroll
  for (int kb = 0; kb < KBLKS; ++kb) {
    a[kb][0] = *(const v16bf*)(s_a + ((kb * 2 + 0) * 32 + l) * 16);
    a[kb][1] = *(const v16bf*)(s_a + ((kb * 2 + 1) * 32 + l) * 16);
  }
  float f2v[8];
#pragma unroll
  for (int r = 0; r < 8; ++r) f2v[r] = s_f2[r + 8 * kh];

  const float sA = 15.0f / mdm[0];

  for (int i = 0; i < 16; ++i) {
    const int nt = w + 8 * i;
    const int n  = nt * 16 + m;
    v8f acc = {};
    const unsigned char* bp = bpack + (size_t)nt * (KBLKS * 2048) + (size_t)l * 32;
#pragma unroll
    for (int kb = 0; kb < KBLKS; ++kb) {
      v16bf bh = *(const v16bf*)(bp);
      v16bf bl = *(const v16bf*)(bp + 1024);
      // f*e ~= hi*hi + hi*lo + lo*hi   (error-compensated bf16)
      acc = __builtin_amdgcn_wmma_f32_16x16x32_bf16(false, a[kb][0], false, bh, (short)0, acc, false, false);
      acc = __builtin_amdgcn_wmma_f32_16x16x32_bf16(false, a[kb][0], false, bl, (short)0, acc, false, false);
      acc = __builtin_amdgcn_wmma_f32_16x16x32_bf16(false, a[kb][1], false, bh, (short)0, acc, false, false);
      bp += 2048;
    }
    const float e2n = e2g[n];
#pragma unroll
    for (int r = 0; r < 8; ++r) {
      float d = f2v[r] + e2n - 2.0f * acc[r];
      s_d[(r + 8 * kh) * ROWSTRIDE + n] = d;
    }
  }
  __syncthreads();

  // ---- Phase C: per-row softmax stats (wave handles 2 rows) ----
#pragma unroll
  for (int rr = 2 * w; rr <= 2 * w + 1; ++rr) {
    float* row = s_d + rr * ROWSTRIDE;
    float mn = 3.4e38f;
    for (int c = l; c < N_CODES; c += 32) mn = fminf(mn, row[c]);
#pragma unroll
    for (int off = 16; off > 0; off >>= 1) mn = fminf(mn, __shfl_xor(mn, off, 32));
    float sm = 0.f;
    for (int c = l; c < N_CODES; c += 32) {
      float e = __expf(sA * (mn - row[c]));   // exp(-s*(d - min)) : max logit normalized
      row[c] = e;
      sm += e;
    }
#pragma unroll
    for (int off = 16; off > 0; off >>= 1) sm += __shfl_xor(sm, off, 32);
    if (l == 0) s_inv[rr] = 1.0f / sm;
  }
  __syncthreads();

  // ---- Phase D: normalize, write codes (float4 along spatial) + bow partials ----
  float inv[16];
#pragma unroll
  for (int i = 0; i < 16; ++i) inv[i] = s_inv[i];
  for (int n = t; n < N_CODES; n += 256) {
    float bs = 0.f, v[16];
#pragma unroll
    for (int i = 0; i < 16; ++i) {
      float c = s_d[i * ROWSTRIDE + n] * inv[i];
      v[i] = c; bs += c;
    }
    float4* op = (float4*)(codes + ((size_t)(b * N_CODES + n)) * POS + pt * 16);
    op[0] = make_float4(v[0],  v[1],  v[2],  v[3]);
    op[1] = make_float4(v[4],  v[5],  v[6],  v[7]);
    op[2] = make_float4(v[8],  v[9],  v[10], v[11]);
    op[3] = make_float4(v[12], v[13], v[14], v[15]);
    bowp[((size_t)(b * PTILES + pt)) * N_CODES + n] = bs;
  }
}

// ---------------- kernel 3: deterministic bow reduce + L1 normalize ----------------
__global__ __launch_bounds__(256)
void k_bow(const float* __restrict__ bowp, float* __restrict__ bow) {
  __shared__ float red[256];
  const int b = blockIdx.x, t = threadIdx.x;
  float v[8]; float loc = 0.f;
#pragma unroll
  for (int j = 0; j < 8; ++j) {
    int n = t + 256 * j;
    float s = 0.f;
    for (int p = 0; p < PTILES; ++p)
      s += bowp[((size_t)(b * PTILES + p)) * N_CODES + n];
    s *= (1.0f / POS);
    v[j] = s;
    loc += fabsf(s);
  }
  red[t] = loc; __syncthreads();
  for (int s = 128; s > 0; s >>= 1) { if (t < s) red[t] += red[t + s]; __syncthreads(); }
  float den = fmaxf(red[0], 1e-12f);
#pragma unroll
  for (int j = 0; j < 8; ++j)
    bow[(size_t)b * N_CODES + t + 256 * j] = v[j] / den;
}

extern "C" void kernel_launch(void* const* d_in, const int* in_sizes, int n_in,
                              void* d_out, int out_size, void* d_ws, size_t ws_size,
                              hipStream_t stream) {
  const float* feat = (const float*)d_in[0];   // (32,256,30,30) f32
  const float* emb  = (const float*)d_in[1];   // (2048,256) f32
  const float* mdm  = (const float*)d_in[2];   // (1,) f32

  float* bow   = (float*)d_out;                          // (32,2048)
  float* codes = (float*)d_out + (size_t)BATCH * N_CODES; // (32,2048,28,28)

  unsigned char* ws    = (unsigned char*)d_ws;
  unsigned char* bpack = ws + WS_BPACK;          // 2 MB packed bf16 B fragments
  float*         e2    = (float*)(ws + WS_E2);   // 8 KB
  float*         bowp  = (float*)(ws + WS_BOWP); // 32*49*2048 f32 partials

  k_e2  <<<N_CODES / 256, 256, 0, stream>>>(emb, e2);
  k_pack<<<NTILES,        256, 0, stream>>>(emb, bpack);

  size_t smem = 16384 + 1024 + 64 + 64 + (size_t)16 * ROWSTRIDE * 4;  // ~145 KB
  k_main<<<dim3(PTILES, BATCH), 256, smem, stream>>>(feat, e2, bpack, mdm, codes, bowp);

  k_bow <<<BATCH, 256, 0, stream>>>(bowp, bow);
}
